// GCNLayer_2010044694696
// MI455X (gfx1250) — matrix operand order repaired and checked by reading, not compiled, not run on hardware
//
#include <hip/hip_runtime.h>
#include <hip/hip_bf16.h>

// ---------------------------------------------------------------------------
// GCN layer: t = X@W + b ; agg = A_spmm(t) ; out = relu(batchnorm(agg))
// X: [N,128] f32, W: [128,128] f32, edges: 1.6M COO.
// GEMM uses v_wmma_f32_16x16x32_bf16 with fragment-ready LDS staging
// (contiguous 32B per lane -> ds_load_b128 pairs, no scalar gathers).
// ---------------------------------------------------------------------------

typedef __bf16 v16bf __attribute__((ext_vector_type(16)));
typedef float  v8f   __attribute__((ext_vector_type(8)));

#define D        128
#define BN_EPS   1e-5f
#define ROWS_PER_BLK 64

// ---------------------------------------------------------------------------
// Kernel 0: zero a float buffer
// ---------------------------------------------------------------------------
__global__ __launch_bounds__(256) void zero_kernel(float* __restrict__ p, long n) {
    long i = (long)blockIdx.x * blockDim.x + threadIdx.x;
    long stride = (long)gridDim.x * blockDim.x;
    for (; i < n; i += stride) p[i] = 0.0f;
}

// ---------------------------------------------------------------------------
// Kernel 1: T = X @ W + b   (WMMA bf16 -> f32)
//   block = 256 threads = 8 waves, covers 64 rows of X.
//   wave w: rowTile rt = w>>1 (16 rows), colTiles = (w&1)*4 .. +4 (64 cols)
//
// LDS holds fragments pre-swizzled into per-lane contiguous 32B chunks:
//   sAf[rt][kk][lane][e] : A frag, e -> k = kk*32 + lhalf*8 + e + (e>=8)*8
//   sBf[kk][nt][lane][e] : B frag, e -> k = kk*32 + lhalf*8 + e + (e>=8)*8, n = nt*16+l16
// so each fragment is a single aligned v16bf load (2x ds_load_b128).
// ---------------------------------------------------------------------------
__global__ __launch_bounds__(256) void gemm_wmma_kernel(
        const float* __restrict__ X, const float* __restrict__ W,
        const float* __restrict__ bias, float* __restrict__ T, int nrows)
{
    __shared__ __bf16 sAf[4][4][32][16];  // [rt][kk][lane][e]  16 KB
    __shared__ __bf16 sBf[4][8][32][16];  // [kk][nt][lane][e]  32 KB

    const int tid     = threadIdx.x;
    const int rowbase = blockIdx.x * ROWS_PER_BLK;

    // cooperative fill: A fragments (f32 -> bf16), 8192 elements
    for (int j = tid; j < 4 * 4 * 32 * 16; j += 256) {
        const int e    = j & 15;
        const int lane = (j >> 4) & 31;
        const int kk   = (j >> 9) & 3;
        const int rt   = (j >> 11) & 3;
        const int row  = rowbase + rt * 16 + (lane & 15);
        const int k    = kk * 32 + (lane >> 4) * 8 + e + ((e >> 3) * 8);
        const float v  = (row < nrows) ? X[(long)row * D + k] : 0.0f;
        ((__bf16*)sAf)[j] = (__bf16)v;
    }
    // cooperative fill: B fragments (f32 -> bf16), 16384 elements (all of W)
    for (int j = tid; j < 4 * 8 * 32 * 16; j += 256) {
        const int e    = j & 15;
        const int lane = (j >> 4) & 31;
        const int nt   = (j >> 9) & 7;
        const int kk   = (j >> 12) & 3;
        const int k    = kk * 32 + (lane >> 4) * 8 + e + ((e >> 3) * 8);
        const int n    = nt * 16 + (lane & 15);
        ((__bf16*)sBf)[j] = (__bf16)W[k * D + n];
    }
    __syncthreads();

    const int wv    = tid >> 5;
    const int lane  = tid & 31;
    const int l16   = lane & 15;
    const int lhalf = lane >> 4;
    const int rt    = wv >> 1;            // row tile 0..3
    const int ct0   = (wv & 1) * 4;       // first col tile (0 or 4)

    v8f acc[4];
    #pragma unroll
    for (int t = 0; t < 4; ++t) acc[t] = (v8f){0,0,0,0,0,0,0,0};

    #pragma unroll
    for (int kk = 0; kk < 4; ++kk) {      // K chunks of 32
        const v16bf afrag = *(const v16bf*)&sAf[rt][kk][lane][0];
        #pragma unroll
        for (int t = 0; t < 4; ++t) {
            const v16bf bfrag = *(const v16bf*)&sBf[kk][ct0 + t][lane][0];
            acc[t] = __builtin_amdgcn_wmma_f32_16x16x32_bf16(
                false, afrag, false, bfrag, (short)0, acc[t], false, false);
        }
    }

    // store C tiles: VGPR r -> row (rt*16 + r + lhalf*8), col = lane&15
    #pragma unroll
    for (int t = 0; t < 4; ++t) {
        const int gcol = (ct0 + t) * 16 + l16;
        const float bb = bias[gcol];
        #pragma unroll
        for (int r = 0; r < 8; ++r) {
            const int grow = rowbase + rt * 16 + r + lhalf * 8;
            if (grow < nrows) T[(long)grow * D + gcol] = acc[t][r] + bb;
        }
    }
}

// ---------------------------------------------------------------------------
// Kernel 2: COO SpMM scatter: agg[row] += val * T[col]
//   one wave per edge, lane handles a float4 (features 4*lane .. 4*lane+3)
// ---------------------------------------------------------------------------
__global__ __launch_bounds__(256) void spmm_kernel(
        const int* __restrict__ rows, const int* __restrict__ cols,
        const float* __restrict__ vals, const float* __restrict__ T,
        float* __restrict__ agg, int nedges)
{
    const int e = blockIdx.x * (blockDim.x >> 5) + (threadIdx.x >> 5);
    if (e >= nedges) return;
    const int lane = threadIdx.x & 31;
    const int r = rows[e];
    const int c = cols[e];
    const float v = vals[e];

    const float4 m = ((const float4*)(T + (long)c * D))[lane];
    float* out = agg + (long)r * D + lane * 4;
    atomicAdd(out + 0, v * m.x);
    atomicAdd(out + 1, v * m.y);
    atomicAdd(out + 2, v * m.z);
    atomicAdd(out + 3, v * m.w);
}

// ---------------------------------------------------------------------------
// Kernel 3: per-column sum / sum-of-squares (128 threads/block, col = tid)
// ---------------------------------------------------------------------------
__global__ __launch_bounds__(128) void bn_stats_kernel(
        const float* __restrict__ agg, float* __restrict__ stats, int nrows)
{
    const int col = threadIdx.x;
    float s = 0.0f, q = 0.0f;
    for (int r = blockIdx.x; r < nrows; r += gridDim.x) {
        float v = agg[(long)r * D + col];
        s += v;
        q += v * v;
    }
    atomicAdd(&stats[col], s);
    atomicAdd(&stats[D + col], q);
}

// ---------------------------------------------------------------------------
// Kernel 4: in-place batchnorm + relu on agg (== d_out)
// ---------------------------------------------------------------------------
__global__ __launch_bounds__(256) void bn_relu_kernel(
        float* __restrict__ agg, const float* __restrict__ stats,
        const float* __restrict__ gamma, const float* __restrict__ beta,
        int nrows)
{
    long idx = (long)blockIdx.x * blockDim.x + threadIdx.x;
    long n = (long)nrows * D;
    if (idx >= n) return;
    const int col = (int)(idx & (D - 1));
    const float invN = 1.0f / (float)nrows;
    const float mean = stats[col] * invN;
    const float var  = stats[D + col] * invN - mean * mean;
    const float o = (agg[idx] - mean) * rsqrtf(var + BN_EPS) * gamma[col] + beta[col];
    agg[idx] = fmaxf(o, 0.0f);
}

// ---------------------------------------------------------------------------
// launch
// ---------------------------------------------------------------------------
extern "C" void kernel_launch(void* const* d_in, const int* in_sizes, int n_in,
                              void* d_out, int out_size, void* d_ws, size_t ws_size,
                              hipStream_t stream)
{
    const float* X     = (const float*)d_in[0];
    const int*   rows  = (const int*)  d_in[1];
    const int*   cols  = (const int*)  d_in[2];
    const float* vals  = (const float*)d_in[3];
    const float* W     = (const float*)d_in[4];
    const float* bias  = (const float*)d_in[5];
    const float* gamma = (const float*)d_in[6];
    const float* beta  = (const float*)d_in[7];

    const int nrows  = in_sizes[0] / D;
    const int nedges = in_sizes[1];

    // workspace: T = [nrows*D] f32, stats = 2*D f32
    float* T     = (float*)d_ws;
    float* stats = T + (long)nrows * D;
    float* agg   = (float*)d_out;   // d_out doubles as the aggregation buffer

    // zero agg (d_out is poisoned) and stats
    {
        long n = (long)nrows * D;
        int blocks = (int)((n + 255) / 256);
        if (blocks > 4096) blocks = 4096;
        zero_kernel<<<blocks, 256, 0, stream>>>(agg, n);
        zero_kernel<<<1, 256, 0, stream>>>(stats, 2 * D);
    }

    // GEMM: T = X @ W + b
    {
        int blocks = (nrows + ROWS_PER_BLK - 1) / ROWS_PER_BLK;
        gemm_wmma_kernel<<<blocks, 256, 0, stream>>>(X, W, bias, T, nrows);
    }

    // SpMM scatter with f32 atomics
    {
        int wavesPerBlock = 256 / 32;
        int blocks = (nedges + wavesPerBlock - 1) / wavesPerBlock;
        spmm_kernel<<<blocks, 256, 0, stream>>>(rows, cols, vals, T, agg, nedges);
    }

    // batchnorm statistics
    bn_stats_kernel<<<512, 128, 0, stream>>>(agg, stats, nrows);

    // normalize + relu in place
    {
        long n = (long)nrows * D;
        int blocks = (int)((n + 255) / 256);
        bn_relu_kernel<<<blocks, 256, 0, stream>>>(agg, stats, gamma, beta, nrows);
    }
}
